// PCIA_25701084299770
// MI455X (gfx1250) — compile-verified, matrix operand order checked by reading, not run on hardware
//
#include <hip/hip_runtime.h>
#include <math.h>

#define WAYS   5
#define NS     5      // way*shot
#define NQ     75     // way*query
#define BTOT   80
#define DIMS   1024
#define NPTS   1024
#define KNN    64     // SPFKNN
#define KS     16     // SPFKS
#define CIFK   15
#define CIFH   64

typedef __attribute__((ext_vector_type(2)))  float     v2f;
typedef __attribute__((ext_vector_type(8)))  float     v8f;
typedef __attribute__((ext_vector_type(16))) _Float16  v16h;

__device__ inline v8f zero8() {
  v8f z;
#pragma unroll
  for (int i = 0; i < 8; ++i) z[i] = 0.f;
  return z;
}
__device__ inline float leaky(float x) { return x > 0.f ? x : 0.2f * x; }

// ---- workspace layout (float offsets) ----
constexpr size_t OFF_G     = 0;          // 80*1024
constexpr size_t OFF_GINV  = 81920;      // 80
constexpr size_t OFF_PN2   = 82048;      // 80*1024
constexpr size_t OFF_COS   = 163968;     // 80*1024
constexpr size_t OFF_CIDX  = 245888;     // 80*64 (int)
constexpr size_t OFF_C2    = 251008;     // 80*64
constexpr size_t OFF_X1    = 256128;     // 80*1024
constexpr size_t OFF_SMETA = 338048;     // 1024
constexpr size_t OFF_X2    = 339072;     // 80*1024
constexpr size_t OFF_RN2   = 420992;     // 80
constexpr size_t OFF_SQATT = 421120;     // 5*1024
constexpr size_t OFF_QSATT = 426240;     // 75*1024
constexpr size_t OFF_IDXSQ = 503040;     // 5*15 (int)
constexpr size_t OFF_IDXQS = 503168;     // 75*5 (int)
constexpr size_t OFF_SATT  = 503552;     // 5*1024
constexpr size_t OFF_QATT  = 508672;     // 75*1024
constexpr size_t OFF_S3    = 585472;     // 5*1024
constexpr size_t OFF_RN3   = 590592;     // 80
constexpr size_t OFF_IDX15 = 590720;     // 5*15 (int)
constexpr size_t OFF_LIDX  = 590848;     // 80*64*16 (int)
constexpr size_t OFF_CFEA  = 672768;     // 80*64*1024
constexpr size_t OFF_D2    = 5915648;    // 80*64*1024
constexpr size_t OFF_QH    = 11158528;   // 80*1024*32 f16 (1310720 floats)
constexpr size_t OFF_KH    = 12469248;
constexpr size_t OFF_VH    = 13779968;   // end: 15090688 floats (~60.4 MB)

// ---- d_out layout (floats) ----
constexpr size_t DOUT_S4   = 0;          // 5*1024
constexpr size_t DOUT_Q3   = 5120;       // 75*1024
constexpr size_t DOUT_CIDX = 81920;      // 80*64
constexpr size_t DOUT_LIDX = 87040;      // 80*64*16

// ============ K0: g_fea = concat(s, q); 1/||g|| ============
__global__ void pcia_prep(const float* s, const float* q, float* ws) {
  int b = blockIdx.x, t = threadIdx.x;
  const float* src = (b < NS) ? (s + b * DIMS) : (q + (b - NS) * DIMS);
  float ss = 0.f;
  for (int d = t; d < DIMS; d += 256) { float v = src[d]; ws[OFF_G + b * DIMS + d] = v; ss += v * v; }
  __shared__ float red[256];
  red[t] = ss; __syncthreads();
  for (int o = 128; o > 0; o >>= 1) { if (t < o) red[t] += red[t + o]; __syncthreads(); }
  if (t == 0) ws[OFF_GINV + b] = 1.f / fmaxf(sqrtf(red[0]), 1e-12f);
}

// ============ K1: point norms + cosine (single streaming pass over p_fea) ============
__global__ void pcia_cospn(const float* p, float* ws) {
  int b = blockIdx.x, n = blockIdx.y * 256 + threadIdx.x;
  __shared__ float gs[DIMS];
  for (int d = threadIdx.x; d < DIMS; d += 256) gs[d] = ws[OFF_G + b * DIMS + d];
  __syncthreads();
  const float* pb = p + (size_t)b * DIMS * NPTS;
  float dot = 0.f, ss = 0.f;
  for (int d = 0; d < DIMS; ++d) { float v = pb[(size_t)d * NPTS + n]; dot += gs[d] * v; ss += v * v; }
  ws[OFF_PN2 + b * NPTS + n] = ss;
  float pin = 1.f / fmaxf(sqrtf(ss), 1e-12f);
  ws[OFF_COS + b * NPTS + n] = 0.5f * dot * ws[OFF_GINV + b] * pin + 0.5f;
}

// ============ K2: stable top-64 of cos per row ============
__global__ void pcia_top64(float* ws, float* dout) {
  int b = blockIdx.x, t = threadIdx.x;
  __shared__ float vals[NPTS];
  __shared__ float rv[256]; __shared__ int ri[256];
  for (int n = t; n < NPTS; n += 256) vals[n] = ws[OFF_COS + b * NPTS + n];
  __syncthreads();
  for (int it = 0; it < KNN; ++it) {
    float bv = -__builtin_inff(); int bi = 0;
    for (int n = t; n < NPTS; n += 256) { float v = vals[n]; if (v > bv || (v == bv && n < bi)) { bv = v; bi = n; } }
    rv[t] = bv; ri[t] = bi; __syncthreads();
    for (int o = 128; o > 0; o >>= 1) {
      if (t < o) { float v2 = rv[t + o]; int i2 = ri[t + o];
        if (v2 > rv[t] || (v2 == rv[t] && i2 < ri[t])) { rv[t] = v2; ri[t] = i2; } }
      __syncthreads();
    }
    if (t == 0) { int idx = ri[0];
      ((int*)ws)[OFF_CIDX + b * KNN + it] = idx;
      dout[DOUT_CIDX + b * KNN + it] = (float)idx;
      vals[idx] = -__builtin_inff(); }
    __syncthreads();
  }
}

// ============ K3: gather center features (columns of p_fea) + their norms ============
__global__ void pcia_gatherc(const float* p, float* ws) {
  int b = blockIdx.x, k = blockIdx.y, t = threadIdx.x;
  int idx = ((int*)ws)[OFF_CIDX + b * KNN + k];
  const float* pb = p + (size_t)b * DIMS * NPTS;
  for (int d = t; d < DIMS; d += 256)
    ws[OFF_CFEA + (size_t)(b * KNN + k) * DIMS + d] = pb[(size_t)d * NPTS + idx];
  if (t == 0) ws[OFF_C2 + b * KNN + k] = ws[OFF_PN2 + b * NPTS + idx];
}

// ============ K4: d2 = |c|^2 + |p|^2 - 2 c·p  via V_WMMA_F32_16X16X4_F32 ============
// grid (80, 16), block 128 (4 waves). Wave w owns rows 16w..16w+15, block owns 64 cols.
__global__ void pcia_d2gemm(const float* p, float* ws) {
  int b = blockIdx.x, n0 = blockIdx.y * 64;
  int lane = threadIdx.x & 31, w = threadIdx.x >> 5;
  int lm = lane & 15, kb = (lane < 16) ? 0 : 2;
  int m0 = w * 16;
  const float* A  = ws + OFF_CFEA + (size_t)b * KNN * DIMS;
  const float* Bm = p + (size_t)b * DIMS * NPTS;
  v8f acc0 = zero8(), acc1 = zero8(), acc2 = zero8(), acc3 = zero8();
  for (int d0 = 0; d0 < DIMS; d0 += 4) {
    v2f a; a.x = A[(size_t)(m0 + lm) * DIMS + d0 + kb];
           a.y = A[(size_t)(m0 + lm) * DIMS + d0 + kb + 1];
    const float* r0 = Bm + (size_t)(d0 + kb) * NPTS;
    const float* r1 = Bm + (size_t)(d0 + kb + 1) * NPTS;
    v2f b0, b1, b2, b3;
    b0.x = r0[n0 + lm];      b0.y = r1[n0 + lm];
    b1.x = r0[n0 + 16 + lm]; b1.y = r1[n0 + 16 + lm];
    b2.x = r0[n0 + 32 + lm]; b2.y = r1[n0 + 32 + lm];
    b3.x = r0[n0 + 48 + lm]; b3.y = r1[n0 + 48 + lm];
    acc0 = __builtin_amdgcn_wmma_f32_16x16x4_f32(false, a, false, b0, (short)0, acc0, false, false);
    acc1 = __builtin_amdgcn_wmma_f32_16x16x4_f32(false, a, false, b1, (short)0, acc1, false, false);
    acc2 = __builtin_amdgcn_wmma_f32_16x16x4_f32(false, a, false, b2, (short)0, acc2, false, false);
    acc3 = __builtin_amdgcn_wmma_f32_16x16x4_f32(false, a, false, b3, (short)0, acc3, false, false);
  }
  int rbase = (lane < 16) ? 0 : 8;
  v8f accs[4] = {acc0, acc1, acc2, acc3};
#pragma unroll
  for (int tI = 0; tI < 4; ++tI) {
    int n = n0 + tI * 16 + lm;
    float pn = ws[OFF_PN2 + b * NPTS + n];
#pragma unroll
    for (int r = 0; r < 8; ++r) {
      int m = m0 + r + rbase;
      float c2 = ws[OFF_C2 + b * KNN + m];
      ws[OFF_D2 + (size_t)(b * KNN + m) * NPTS + n] = c2 + pn - 2.f * accs[tI][r];
    }
  }
}

// ============ K5: top-16 smallest d2 per (b,k) row ============
__global__ void pcia_top16(float* ws, float* dout) {
  int row = blockIdx.x, t = threadIdx.x;   // row = b*64+k
  __shared__ float vals[NPTS];
  __shared__ float rv[256]; __shared__ int ri[256];
  const float* src = ws + OFF_D2 + (size_t)row * NPTS;
  for (int n = t; n < NPTS; n += 256) vals[n] = src[n];
  __syncthreads();
  for (int it = 0; it < KS; ++it) {
    float bv = __builtin_inff(); int bi = 0;
    for (int n = t; n < NPTS; n += 256) { float v = vals[n]; if (v < bv || (v == bv && n < bi)) { bv = v; bi = n; } }
    rv[t] = bv; ri[t] = bi; __syncthreads();
    for (int o = 128; o > 0; o >>= 1) {
      if (t < o) { if (rv[t + o] < rv[t] || (rv[t + o] == rv[t] && ri[t + o] < ri[t])) { rv[t] = rv[t + o]; ri[t] = ri[t + o]; } }
      __syncthreads();
    }
    if (t == 0) { int idx = ri[0];
      ((int*)ws)[OFF_LIDX + row * KS + it] = idx;
      dout[DOUT_LIDX + row * KS + it] = (float)idx;
      vals[idx] = __builtin_inff(); }
    __syncthreads();
  }
}

// ============ K6: conv11 over grouped neighbors + max over k; x1 = (spf+g)/2 ============
__global__ void pcia_spf(const float* p, const float* w11, float* ws) {
  int b = blockIdx.x, d = blockIdx.y * 256 + threadIdx.x;
  __shared__ int   sl[KNN * KS];
  __shared__ float w11s[KS + 1];
  for (int i = threadIdx.x; i < KNN * KS; i += 256) sl[i] = ((int*)ws)[OFF_LIDX + b * KNN * KS + i];
  if (threadIdx.x < KS + 1) w11s[threadIdx.x] = w11[threadIdx.x];
  __syncthreads();
  const float* prow = p + (size_t)b * DIMS * NPTS + (size_t)d * NPTS;
  float g = ws[OFF_G + b * DIMS + d];
  float base = w11s[KS] * g;
  float best = -__builtin_inff();
  for (int k = 0; k < KNN; ++k) {
    float acc = base;
#pragma unroll
    for (int c = 0; c < KS; ++c) acc += w11s[c] * prow[sl[k * KS + c]];
    best = fmaxf(best, acc);
  }
  ws[OFF_X1 + b * DIMS + d] = 0.5f * (best + g);
}

// ============ K7: s_meta = leaky(w_fuse1 · s1) ============
__global__ void pcia_smeta(const float* wf1, float* ws) {
  int d = blockIdx.x * 256 + threadIdx.x;
  float a = 0.f;
  for (int w = 0; w < WAYS; ++w) a += wf1[w] * ws[OFF_X1 + w * DIMS + d];
  ws[OFF_SMETA + d] = leaky(a);
}

// ============ K8: build Q,K,V (fp16) for the 32-dim fuse attention ============
__global__ void pcia_qkv(const float* wf2, const float* Wfc1, const float* Wfc2,
                         const float* Wfc3, float* ws) {
  __shared__ float W1[1024], W2[1024], W3[1024], F2[64];
  int t = threadIdx.x;
  for (int i = t; i < 1024; i += 256) { W1[i] = Wfc1[i]; W2[i] = Wfc2[i]; W3[i] = Wfc3[i]; }
  if (t < 64) F2[t] = wf2[t];
  __syncthreads();
  int b = blockIdx.x, d = blockIdx.y * 256 + t;
  float x = ws[OFF_X1 + b * DIMS + d], sm = ws[OFF_SMETA + d];
  float xf[32];
#pragma unroll
  for (int c = 0; c < 32; ++c) xf[c] = leaky(F2[c * 2] * x + F2[c * 2 + 1] * sm);
  _Float16* qh = (_Float16*)(ws + OFF_QH);
  _Float16* kh = (_Float16*)(ws + OFF_KH);
  _Float16* vh = (_Float16*)(ws + OFF_VH);
  size_t o0 = ((size_t)b * DIMS + d) * 32;
  for (int o = 0; o < 32; ++o) {
    float aq = 0.f, ak = 0.f, av = 0.f;
#pragma unroll
    for (int c = 0; c < 32; ++c) { float xv = xf[c]; aq += W1[o * 32 + c] * xv; ak += W2[o * 32 + c] * xv; av += W3[o * 32 + c] * xv; }
    qh[o0 + o] = (_Float16)aq; kh[o0 + o] = (_Float16)ak; vh[o0 + o] = (_Float16)av;
  }
}

// ============ K9: fuse attention, one wave per 16 query rows; WMMA f16 QK^T & AV ============
__global__ void pcia_attn(const float* wf3, float* ws) {
  extern __shared__ float lds[];                 // 16*1024 logits (64 KB)
  int b = blockIdx.x, q0 = blockIdx.y * 16;
  int lane = threadIdx.x;
  int lm = lane & 15, hi = lane >> 4;
  const _Float16* qh = (const _Float16*)(ws + OFF_QH);
  const _Float16* kh = (const _Float16*)(ws + OFF_KH);
  const _Float16* vh = (const _Float16*)(ws + OFF_VH);
  v16h aq;
  const _Float16* qrow = qh + ((size_t)b * DIMS + q0 + lm) * 32;
#pragma unroll
  for (int j = 0; j < 16; ++j) { int c = (j >> 3) * 16 + hi * 8 + (j & 7); aq[j] = qrow[c]; }
  const float scale = 0.17677669529663687f;      // 1/sqrt(32)
  for (int e0 = 0; e0 < NPTS; e0 += 16) {
    v16h bk;
    const _Float16* krow = kh + ((size_t)b * DIMS + e0 + lm) * 32;
#pragma unroll
    for (int j = 0; j < 16; ++j) { int c = (j >> 3) * 16 + hi * 8 + (j & 7); bk[j] = krow[c]; }
    v8f cc = zero8();
    cc = __builtin_amdgcn_wmma_f32_16x16x32_f16(false, aq, false, bk, (short)0, cc, false, false);
#pragma unroll
    for (int r = 0; r < 8; ++r) lds[(r + hi * 8) * NPTS + e0 + lm] = cc[r] * scale;
  }
  __syncthreads();
  for (int r = 0; r < 16; ++r) {                 // row softmax (fp32)
    float mx = -__builtin_inff();
    for (int e = lane; e < NPTS; e += 32) mx = fmaxf(mx, lds[r * NPTS + e]);
    for (int o = 16; o > 0; o >>= 1) mx = fmaxf(mx, __shfl_xor(mx, o, 32));
    float sum = 0.f;
    for (int e = lane; e < NPTS; e += 32) { float ex = __expf(lds[r * NPTS + e] - mx); lds[r * NPTS + e] = ex; sum += ex; }
    for (int o = 16; o > 0; o >>= 1) sum += __shfl_xor(sum, o, 32);
    float inv = 1.f / sum;
    for (int e = lane; e < NPTS; e += 32) lds[r * NPTS + e] *= inv;
  }
  __syncthreads();
  v8f acc0 = zero8(), acc1 = zero8();
  for (int e0 = 0; e0 < NPTS; e0 += 32) {
    v16h am, bv0, bv1;
#pragma unroll
    for (int j = 0; j < 16; ++j) {
      int e = e0 + (j >> 3) * 16 + hi * 8 + (j & 7);
      am[j] = (_Float16)lds[lm * NPTS + e];
      const _Float16* vr = vh + ((size_t)b * DIMS + e) * 32;
      bv0[j] = vr[lm]; bv1[j] = vr[16 + lm];
    }
    acc0 = __builtin_amdgcn_wmma_f32_16x16x32_f16(false, am, false, bv0, (short)0, acc0, false, false);
    acc1 = __builtin_amdgcn_wmma_f32_16x16x32_f16(false, am, false, bv1, (short)0, acc1, false, false);
  }
  __syncthreads();
#pragma unroll
  for (int r = 0; r < 8; ++r) { int row = r + hi * 8;
    lds[row * 32 + lm] = acc0[r]; lds[row * 32 + 16 + lm] = acc1[r]; }
  __syncthreads();
  if (lane < 16) {
    float a = 0.f;
    for (int o = 0; o < 32; ++o) a += wf3[o] * lds[lane * 32 + o];
    size_t id = (size_t)b * DIMS + q0 + lane;
    ws[OFF_X2 + id] = 0.5f * (leaky(a) + ws[OFF_X1 + id]);
  }
}

// ============ generic: inverse L2 norm per 1024-row ============
__global__ void pcia_rownorm(const float* src, float* dst) {
  int row = blockIdx.x, t = threadIdx.x;
  __shared__ float red[256];
  float ss = 0.f;
  for (int d = t; d < DIMS; d += 256) { float v = src[(size_t)row * DIMS + d]; ss += v * v; }
  red[t] = ss; __syncthreads();
  for (int o = 128; o > 0; o >>= 1) { if (t < o) red[t] += red[t + o]; __syncthreads(); }
  if (t == 0) dst[row] = 1.f / fmaxf(sqrtf(red[0]), 1e-12f);
}

// ============ K11: s-side cross attention + top-15 cosine indices ============
__global__ void pcia_sq(float* ws) {
  int i = blockIdx.x, t = threadIdx.x;
  __shared__ float dots[NQ], aw[NQ];
  const float* s2 = ws + OFF_X2;
  const float* q2 = ws + OFF_X2 + (size_t)NS * DIMS;
  if (t < NQ) { float a = 0.f; for (int d = 0; d < DIMS; ++d) a += s2[i * DIMS + d] * q2[(size_t)t * DIMS + d]; dots[t] = a; }
  __syncthreads();
  if (t == 0) {
    const float sc = 1.f / 32.f;
    float mx = -__builtin_inff();
    for (int j = 0; j < NQ; ++j) mx = fmaxf(mx, dots[j] * sc);
    float sum = 0.f;
    for (int j = 0; j < NQ; ++j) { aw[j] = __expf(dots[j] * sc - mx); sum += aw[j]; }
    for (int j = 0; j < NQ; ++j) aw[j] /= sum;
    float rs = ws[OFF_RN2 + i], cv[NQ];
    for (int j = 0; j < NQ; ++j) cv[j] = dots[j] * rs * ws[OFF_RN2 + NS + j];
    for (int it = 0; it < CIFK; ++it) {
      float bv = -__builtin_inff(); int bi = 0;
      for (int j = 0; j < NQ; ++j) if (cv[j] > bv) { bv = cv[j]; bi = j; }
      ((int*)ws)[OFF_IDXSQ + i * CIFK + it] = bi; cv[bi] = -__builtin_inff();
    }
  }
  __syncthreads();
  for (int d = t; d < DIMS; d += 256) {
    float a = 0.f;
    for (int j = 0; j < NQ; ++j) a += aw[j] * q2[(size_t)j * DIMS + d];
    ws[OFF_SQATT + i * DIMS + d] = a;
  }
}

// ============ K12: q-side cross attention + top-5 cosine indices ============
__global__ void pcia_qs(float* ws) {
  int jq = blockIdx.x, t = threadIdx.x;
  __shared__ float dots[WAYS], aw[WAYS];
  const float* s2 = ws + OFF_X2;
  const float* q2 = ws + OFF_X2 + (size_t)NS * DIMS;
  if (t < WAYS) { float a = 0.f; for (int d = 0; d < DIMS; ++d) a += q2[(size_t)jq * DIMS + d] * s2[t * DIMS + d]; dots[t] = a; }
  __syncthreads();
  if (t == 0) {
    const float sc = 1.f / 32.f;
    float mx = -__builtin_inff();
    for (int w = 0; w < WAYS; ++w) mx = fmaxf(mx, dots[w] * sc);
    float sum = 0.f;
    for (int w = 0; w < WAYS; ++w) { aw[w] = __expf(dots[w] * sc - mx); sum += aw[w]; }
    for (int w = 0; w < WAYS; ++w) aw[w] /= sum;
    float rq = ws[OFF_RN2 + NS + jq], cv[WAYS];
    for (int w = 0; w < WAYS; ++w) cv[w] = dots[w] * rq * ws[OFF_RN2 + w];
    for (int it = 0; it < WAYS; ++it) {
      float bv = -__builtin_inff(); int bi = 0;
      for (int w = 0; w < WAYS; ++w) if (cv[w] > bv) { bv = cv[w]; bi = w; }
      ((int*)ws)[OFF_IDXQS + jq * WAYS + it] = bi; cv[bi] = -__builtin_inff();
    }
  }
  __syncthreads();
  for (int d = t; d < DIMS; d += 256) {
    float a = 0.f;
    for (int w = 0; w < WAYS; ++w) a += aw[w] * s2[w * DIMS + d];
    ws[OFF_QSATT + jq * DIMS + d] = a;
  }
}

// ============ K13: CIF s-branch (convbn w1 -> convbn w3 -> softmax over 16) ============
__global__ void pcia_cif_s(const float* w1, const float* g1, const float* b1,
                           const float* w3, const float* g3, const float* b3, float* ws) {
  __shared__ float W1s[CIFH * 16], W3s[16 * CIFH], G1s[CIFH], B1s[CIFH], G3s[16], B3s[16];
  __shared__ int idxs[CIFK];
  int t = threadIdx.x;
  for (int i = t; i < CIFH * 16; i += 256) { W1s[i] = w1[i]; W3s[i] = w3[i]; }
  if (t < CIFH) { G1s[t] = g1[t]; B1s[t] = b1[t]; }
  if (t < 16)   { G3s[t] = g3[t]; B3s[t] = b3[t]; }
  int i = blockIdx.x;
  if (t < CIFK) idxs[t] = ((int*)ws)[OFF_IDXSQ + i * CIFK + t];
  __syncthreads();
  int d = blockIdx.y * 256 + t;
  const float inv = rsqrtf(1.f + 1e-5f);
  float v[16];
  v[0] = ws[OFF_X2 + i * DIMS + d];
#pragma unroll
  for (int c = 0; c < CIFK; ++c) v[1 + c] = ws[OFF_X2 + (size_t)(NS + idxs[c]) * DIMS + d];
  float h[CIFH];
  for (int o = 0; o < CIFH; ++o) {
    float a = 0.f;
#pragma unroll
    for (int c = 0; c < 16; ++c) a += W1s[o * 16 + c] * v[c];
    h[o] = leaky(a * (G1s[o] * inv) + B1s[o]);
  }
  float y[16], mx = -__builtin_inff();
  for (int c = 0; c < 16; ++c) {
    float a = 0.f;
    for (int o = 0; o < CIFH; ++o) a += W3s[c * CIFH + o] * h[o];
    y[c] = leaky(a * (G3s[c] * inv) + B3s[c]);
    mx = fmaxf(mx, y[c]);
  }
  float sum = 0.f;
#pragma unroll
  for (int c = 0; c < 16; ++c) { y[c] = __expf(y[c] - mx); sum += y[c]; }
  float out = 0.f;
#pragma unroll
  for (int c = 0; c < 16; ++c) out += (y[c] / sum) * v[c];
  ws[OFF_SATT + i * DIMS + d] = out;
}

// ============ K14: CIF q-branch (convbn w2 -> convbn w4 -> softmax over 6) ============
__global__ void pcia_cif_q(const float* w2, const float* g2, const float* b2,
                           const float* w4, const float* g4, const float* b4, float* ws) {
  __shared__ float W2s[CIFH * 6], W4s[6 * CIFH], G2s[CIFH], B2s[CIFH], G4s[6], B4s[6];
  __shared__ int idxs[WAYS];
  int t = threadIdx.x;
  for (int i = t; i < CIFH * 6; i += 256) { W2s[i] = w2[i]; W4s[i] = w4[i]; }
  if (t < CIFH) { G2s[t] = g2[t]; B2s[t] = b2[t]; }
  if (t < 6)    { G4s[t] = g4[t]; B4s[t] = b4[t]; }
  int jq = blockIdx.x;
  if (t < WAYS) idxs[t] = ((int*)ws)[OFF_IDXQS + jq * WAYS + t];
  __syncthreads();
  int d = blockIdx.y * 256 + t;
  const float inv = rsqrtf(1.f + 1e-5f);
  float v[6];
#pragma unroll
  for (int w = 0; w < WAYS; ++w) v[w] = ws[OFF_X2 + (size_t)idxs[w] * DIMS + d];
  v[5] = ws[OFF_X2 + (size_t)(NS + jq) * DIMS + d];
  float h[CIFH];
  for (int o = 0; o < CIFH; ++o) {
    float a = 0.f;
#pragma unroll
    for (int c = 0; c < 6; ++c) a += W2s[o * 6 + c] * v[c];
    h[o] = leaky(a * (G2s[o] * inv) + B2s[o]);
  }
  float y[6], mx = -__builtin_inff();
  for (int c = 0; c < 6; ++c) {
    float a = 0.f;
    for (int o = 0; o < CIFH; ++o) a += W4s[c * CIFH + o] * h[o];
    y[c] = leaky(a * (G4s[c] * inv) + B4s[c]);
    mx = fmaxf(mx, y[c]);
  }
  float sum = 0.f;
#pragma unroll
  for (int c = 0; c < 6; ++c) { y[c] = __expf(y[c] - mx); sum += y[c]; }
  float out = 0.f;
#pragma unroll
  for (int c = 0; c < 6; ++c) out += (y[c] / sum) * v[c];
  ws[OFF_QATT + jq * DIMS + d] = out;
}

// ============ K15: s3/q3 ============
__global__ void pcia_x3(float* ws, float* dout) {
  int row = blockIdx.x, d = blockIdx.y * 256 + threadIdx.x;
  if (row < NS) {
    ws[OFF_S3 + row * DIMS + d] =
      (ws[OFF_SATT + row * DIMS + d] + ws[OFF_X2 + row * DIMS + d] + ws[OFF_SQATT + row * DIMS + d]) / 3.f;
  } else {
    int j = row - NS;
    dout[DOUT_Q3 + (size_t)j * DIMS + d] =
      (ws[OFF_QATT + (size_t)j * DIMS + d] + ws[OFF_X2 + (size_t)row * DIMS + d] + ws[OFF_QSATT + (size_t)j * DIMS + d]) / 3.f;
  }
}

// ============ K16: top-15 cosine(s3, q3) ============
__global__ void pcia_idx15(float* ws, const float* dout) {
  int i = blockIdx.x, t = threadIdx.x;
  __shared__ float cv[NQ];
  if (t < NQ) {
    float a = 0.f;
    const float* s3 = ws + OFF_S3 + (size_t)i * DIMS;
    const float* q3 = dout + DOUT_Q3 + (size_t)t * DIMS;
    for (int d = 0; d < DIMS; ++d) a += s3[d] * q3[d];
    cv[t] = a * ws[OFF_RN3 + i] * ws[OFF_RN3 + NS + t];
  }
  __syncthreads();
  if (t == 0) {
    for (int it = 0; it < CIFK; ++it) {
      float bv = -__builtin_inff(); int bi = 0;
      for (int j = 0; j < NQ; ++j) if (cv[j] > bv) { bv = cv[j]; bi = j; }
      ((int*)ws)[OFF_IDX15 + i * CIFK + it] = bi; cv[bi] = -__builtin_inff();
    }
  }
}

// ============ K17: final CIF block -> s4 ============
__global__ void pcia_s4(const float* w5, const float* g5, const float* b5,
                        const float* w6, const float* g6, const float* b6,
                        float* ws, float* dout) {
  __shared__ float W5s[CIFH * 16], W6s[16 * CIFH], G5s[CIFH], B5s[CIFH], G6s[16], B6s[16];
  __shared__ int idxs[CIFK];
  int t = threadIdx.x;
  for (int i = t; i < CIFH * 16; i += 256) { W5s[i] = w5[i]; W6s[i] = w6[i]; }
  if (t < CIFH) { G5s[t] = g5[t]; B5s[t] = b5[t]; }
  if (t < 16)   { G6s[t] = g6[t]; B6s[t] = b6[t]; }
  int i = blockIdx.x;
  if (t < CIFK) idxs[t] = ((int*)ws)[OFF_IDX15 + i * CIFK + t];
  __syncthreads();
  int d = blockIdx.y * 256 + t;
  const float inv = rsqrtf(1.f + 1e-5f);
  float v[16];
  v[0] = ws[OFF_S3 + i * DIMS + d];
#pragma unroll
  for (int c = 0; c < CIFK; ++c) v[1 + c] = dout[DOUT_Q3 + (size_t)idxs[c] * DIMS + d];
  float h[CIFH];
  for (int o = 0; o < CIFH; ++o) {
    float a = 0.f;
#pragma unroll
    for (int c = 0; c < 16; ++c) a += W5s[o * 16 + c] * v[c];
    h[o] = leaky(a * (G5s[o] * inv) + B5s[o]);
  }
  float y[16], mx = -__builtin_inff();
  for (int c = 0; c < 16; ++c) {
    float a = 0.f;
    for (int o = 0; o < CIFH; ++o) a += W6s[c * CIFH + o] * h[o];
    y[c] = leaky(a * (G6s[c] * inv) + B6s[c]);
    mx = fmaxf(mx, y[c]);
  }
  float sum = 0.f;
#pragma unroll
  for (int c = 0; c < 16; ++c) { y[c] = __expf(y[c] - mx); sum += y[c]; }
  float out = 0.f;
#pragma unroll
  for (int c = 0; c < 16; ++c) out += (y[c] / sum) * v[c];
  dout[DOUT_S4 + i * DIMS + d] = 0.5f * (out + v[0]);
}

extern "C" void kernel_launch(void* const* d_in, const int* in_sizes, int n_in,
                              void* d_out, int out_size, void* d_ws, size_t ws_size,
                              hipStream_t stream) {
  (void)in_sizes; (void)n_in; (void)out_size; (void)ws_size;
  const float* s     = (const float*)d_in[0];
  const float* q     = (const float*)d_in[1];
  const float* p     = (const float*)d_in[2];
  // d_in[3] = coords (unused by the reference)
  const float* w11   = (const float*)d_in[4];
  const float* wf1   = (const float*)d_in[5];
  const float* wf2   = (const float*)d_in[6];
  const float* wf3   = (const float*)d_in[7];
  const float* Wfc1  = (const float*)d_in[8];
  const float* Wfc2  = (const float*)d_in[9];
  const float* Wfc3  = (const float*)d_in[10];
  const float* w1 = (const float*)d_in[11]; const float* g1 = (const float*)d_in[12]; const float* b1 = (const float*)d_in[13];
  const float* w2 = (const float*)d_in[14]; const float* g2 = (const float*)d_in[15]; const float* b2 = (const float*)d_in[16];
  const float* w3 = (const float*)d_in[17]; const float* g3 = (const float*)d_in[18]; const float* b3 = (const float*)d_in[19];
  const float* w4 = (const float*)d_in[20]; const float* g4 = (const float*)d_in[21]; const float* b4 = (const float*)d_in[22];
  const float* w5 = (const float*)d_in[23]; const float* g5 = (const float*)d_in[24]; const float* b5 = (const float*)d_in[25];
  const float* w6 = (const float*)d_in[26]; const float* g6 = (const float*)d_in[27]; const float* b6 = (const float*)d_in[28];
  float* ws   = (float*)d_ws;
  float* dout = (float*)d_out;

  pcia_prep   <<<BTOT, 256, 0, stream>>>(s, q, ws);
  pcia_cospn  <<<dim3(BTOT, 4), 256, 0, stream>>>(p, ws);
  pcia_top64  <<<BTOT, 256, 0, stream>>>(ws, dout);
  pcia_gatherc<<<dim3(BTOT, KNN), 256, 0, stream>>>(p, ws);
  pcia_d2gemm <<<dim3(BTOT, 16), 128, 0, stream>>>(p, ws);
  pcia_top16  <<<BTOT * KNN, 256, 0, stream>>>(ws, dout);
  pcia_spf    <<<dim3(BTOT, 4), 256, 0, stream>>>(p, w11, ws);
  pcia_smeta  <<<4, 256, 0, stream>>>(wf1, ws);
  pcia_qkv    <<<dim3(BTOT, 4), 256, 0, stream>>>(wf2, Wfc1, Wfc2, Wfc3, ws);
  pcia_attn   <<<dim3(BTOT, 64), 32, 16 * NPTS * sizeof(float), stream>>>(wf3, ws);
  pcia_rownorm<<<BTOT, 256, 0, stream>>>(ws + OFF_X2, ws + OFF_RN2);
  pcia_sq     <<<NS, 256, 0, stream>>>(ws);
  pcia_qs     <<<NQ, 256, 0, stream>>>(ws);
  pcia_cif_s  <<<dim3(NS, 4), 256, 0, stream>>>(w1, g1, b1, w3, g3, b3, ws);
  pcia_cif_q  <<<dim3(NQ, 4), 256, 0, stream>>>(w2, g2, b2, w4, g4, b4, ws);
  pcia_x3     <<<dim3(BTOT, 4), 256, 0, stream>>>(ws, dout);
  pcia_rownorm<<<NS, 256, 0, stream>>>(ws + OFF_S3, ws + OFF_RN3);
  pcia_rownorm<<<NQ, 256, 0, stream>>>(dout + DOUT_Q3, ws + OFF_RN3 + NS);
  pcia_idx15  <<<NS, 256, 0, stream>>>(ws, dout);
  pcia_s4     <<<dim3(NS, 4), 256, 0, stream>>>(w5, g5, b5, w6, g6, b6, ws, dout);
}
